// MultiHeadAttention_87651692576779
// MI455X (gfx1250) — matrix operand order repaired and checked by reading, not compiled, not run on hardware
//
#include <hip/hip_runtime.h>
#include <hip/hip_bf16.h>

// ---------------------------------------------------------------------------
// CDNA5 (gfx1250) wave32 WMMA types
// ---------------------------------------------------------------------------
typedef __attribute__((ext_vector_type(16))) __bf16 v16bf;
typedef __attribute__((ext_vector_type(8)))  __bf16 v8bf;
typedef __attribute__((ext_vector_type(8)))  float  v8f;

typedef __attribute__((ext_vector_type(4))) unsigned int tdm_g0_t;
typedef __attribute__((ext_vector_type(8))) int          tdm_g1_t;
typedef __attribute__((ext_vector_type(4))) int          tdm_g2_t;

#if __has_builtin(__builtin_amdgcn_tensor_load_to_lds)
#define HAS_TDM 1
#else
#define HAS_TDM 0
#endif

__device__ __forceinline__ unsigned short f32_to_bf16(float f) {
  unsigned u = __float_as_uint(f);
  unsigned r = (u + 0x7FFFu + ((u >> 16) & 1u)) >> 16;   // round-to-nearest-even
  return (unsigned short)r;
}

__device__ __forceinline__ v16bf cat16(v8bf lo, v8bf hi) {
  return __builtin_shufflevector(lo, hi, 0,1,2,3,4,5,6,7,8,9,10,11,12,13,14,15);
}

__device__ __forceinline__ v8f zero_v8f() {
  v8f z;
#pragma unroll
  for (int i = 0; i < 8; i++) z[i] = 0.0f;
  return z;
}

__device__ __forceinline__ v8f wmma_bf16(v16bf a, v16bf b, v8f c) {
  return __builtin_amdgcn_wmma_f32_16x16x32_bf16(false, a, false, b, (short)0, c, false, false);
}

// A-fragment (16-bit, 16x32): lane<16 -> K {0..7,16..23}; lane>=16 -> K {8..15,24..31}
__device__ __forceinline__ v16bf load_a_frag(const unsigned short* row, int koff) {
  v8bf lo = *(const v8bf*)(row + koff);
  v8bf hi = *(const v8bf*)(row + koff + 16);
  return cat16(lo, hi);
}
// B-fragment (16-bit, 32x16) from a [N][K] (transposed) tile: lane<16 -> K {0..7,16..23}; lane>=16 -> K {8..15,24..31}
__device__ __forceinline__ v16bf load_b_frag(const unsigned short* row, int kbase) {
  v8bf lo = *(const v8bf*)(row + kbase);
  v8bf hi = *(const v8bf*)(row + kbase + 16);
  return cat16(lo, hi);
}

// ---------------------------------------------------------------------------
// TDM: 2D bf16 tile DMA global -> LDS with hardware LDS padding.
//   tile_d0 = contiguous elements per row, tile_d1 = rows,
//   row_stride = global row stride (elements), pad codes per D# spec.
// D# packing per CDNA5 ISA ch.8 (group0: count/lds/global/type; group1: sizes).
// This toolchain exposes the 6-arg builtin:
//   (uint32x4 g0, int32x8 g1, int32x4 g2, int32x4 g3, int32x8 pad, i32 cpol)
// ---------------------------------------------------------------------------
#if HAS_TDM
__device__ __forceinline__ void tdm_load_2d_bf16(
    const void* gsrc, unsigned ldsoff,
    unsigned tile_d0, unsigned tile_d1,
    unsigned long long row_stride,
    unsigned pad_interval_code, unsigned pad_amount_code)
{
  unsigned long long ga = (unsigned long long)gsrc;
  tdm_g0_t g0;
  g0[0] = 1u;                                         // count=1 (valid), user mode
  g0[1] = ldsoff;                                     // lds_addr (bytes)
  g0[2] = (unsigned)(ga & 0xFFFFFFFFu);               // global_addr[31:0]
  g0[3] = (unsigned)((ga >> 32) & 0x01FFFFFFu) | (2u << 30);   // addr[56:32] | type=2

  tdm_g1_t g1;
  unsigned w0 = (1u << 16)                            // data_size = 1 -> 2 bytes
              | (1u << 20)                            // pad_enable
              | (pad_interval_code << 22)
              | (pad_amount_code << 25);
  unsigned long long td0 = tile_d0, td1 = tile_d1;    // tensor dims == tile dims
  g1[0] = (int)w0;
  g1[1] = (int)((td0 & 0xFFFFu) << 16);                               // tensor_dim0[15:0]
  g1[2] = (int)(((td0 >> 16) & 0xFFFFu) | ((td1 & 0xFFFFu) << 16));   // td0 hi | td1 lo
  g1[3] = (int)(((td1 >> 16) & 0xFFFFu) | ((tile_d0 & 0xFFFFu) << 16)); // td1 hi | tile_dim0
  g1[4] = (int)(tile_d1 & 0xFFFFu);                                   // tile_dim1 (tile_dim2=0)
  g1[5] = (int)(row_stride & 0xFFFFFFFFu);                            // dim0_stride[31:0]
  g1[6] = (int)((row_stride >> 32) & 0xFFFFu);                        // dim0_stride[47:32]
  g1[7] = 0;

  tdm_g2_t z4 = {0, 0, 0, 0};
  tdm_g1_t z8 = {0, 0, 0, 0, 0, 0, 0, 0};
  __builtin_amdgcn_tensor_load_to_lds(g0, g1, z4, z4, z8, 0);
}
#endif

__device__ __forceinline__ unsigned lds_byte_off(const void* p) {
  // low 32 bits of a flat shared pointer == LDS byte offset (ISA 10.2)
  return (unsigned)(unsigned long long)p;
}

// ---------------------------------------------------------------------------
// Problem constants
// ---------------------------------------------------------------------------
#define BATCH   4
#define SEQ     1024
#define DMODEL  2048
#define NHEADS  16
#define HEADDIM 128
#define MROWS   (BATCH * SEQ)              // 4096
#define QKV_SCALE 0.0883883476483184405f   // 1/sqrt(128)

// ---------------------------------------------------------------------------
// Tiled WMMA GEMM:  D = A(MxK) * W(KxN) + bias, register-prefetch pipelined.
//  EPI==0 : A f32 (x), scatter epilogue -> Q (scaled), K, V^T (bf16)
//  EPI==1 : A bf16 (attn_out), epilogue -> f32 d_out
// ---------------------------------------------------------------------------
template <int EPI>
__global__ __launch_bounds__(256) void gemm_wmma_kernel(
    const float* __restrict__ Af, const unsigned short* __restrict__ Ab,
    const float* __restrict__ W, const float* __restrict__ bias,
    int K, int N,
    unsigned short* __restrict__ qout, unsigned short* __restrict__ kout,
    unsigned short* __restrict__ vtout, float* __restrict__ fout)
{
  constexpr int BM = 128, BN = 128, BK = 32;
  constexpr int ASTR = 40, BSTR = 40;
  __shared__ __align__(16) unsigned short Als[BM * ASTR];   // [m][k]
  __shared__ __align__(16) unsigned short Bls[BN * BSTR];   // [n][k] (W transposed)

  const int tid  = threadIdx.x;
  const int lane = tid & 31;
  const int wave = tid >> 5;
  const int m0 = blockIdx.y * BM;
  const int n0 = blockIdx.x * BN;
  const int wm = (wave >> 1) * 32;
  const int wn = (wave & 1) * 64;

  v8f acc[2][4];
#pragma unroll
  for (int i = 0; i < 2; i++)
#pragma unroll
    for (int j = 0; j < 4; j++) acc[i][j] = zero_v8f();

  const int arow  = tid >> 1;          // A: 128 rows x 2 k-segs of 16
  const int akseg = (tid & 1) * 16;
  const int bk    = tid >> 3;          // B: 32 k x 8 n-segs of 16
  const int bnseg = (tid & 7) * 16;

  const int frow   = lane & 15;
  const int akoff  = (lane < 16) ? 0 : 8;
  const int bkbase = (lane < 16) ? 0 : 8;

  float4 fa[4]; uint4 ua[2]; float4 fb[4];
  auto load_globals = [&](int kk) {
    if (EPI == 0) {
      const float* ap = Af + (size_t)(m0 + arow) * K + kk + akseg;
#pragma unroll
      for (int i = 0; i < 4; i++) fa[i] = ((const float4*)ap)[i];
    } else {
      const unsigned short* ap = Ab + (size_t)(m0 + arow) * K + kk + akseg;
      ua[0] = ((const uint4*)ap)[0];
      ua[1] = ((const uint4*)ap)[1];
    }
    const float* wp = W + (size_t)(kk + bk) * N + n0 + bnseg;
#pragma unroll
    for (int i = 0; i < 4; i++) fb[i] = ((const float4*)wp)[i];
  };

  load_globals(0);

  for (int kk = 0; kk < K; kk += BK) {
    __syncthreads();
    // ---- commit staged registers to LDS ----
    if (EPI == 0) {
      unsigned short hv[16];
#pragma unroll
      for (int i = 0; i < 4; i++) {
        hv[4*i+0] = f32_to_bf16(fa[i].x); hv[4*i+1] = f32_to_bf16(fa[i].y);
        hv[4*i+2] = f32_to_bf16(fa[i].z); hv[4*i+3] = f32_to_bf16(fa[i].w);
      }
      *(uint4*)&Als[arow * ASTR + akseg]     = *(const uint4*)&hv[0];
      *(uint4*)&Als[arow * ASTR + akseg + 8] = *(const uint4*)&hv[8];
    } else {
      *(uint4*)&Als[arow * ASTR + akseg]     = ua[0];
      *(uint4*)&Als[arow * ASTR + akseg + 8] = ua[1];
    }
#pragma unroll
    for (int i = 0; i < 4; i++) {
      Bls[(bnseg + 4*i + 0) * BSTR + bk] = f32_to_bf16(fb[i].x);
      Bls[(bnseg + 4*i + 1) * BSTR + bk] = f32_to_bf16(fb[i].y);
      Bls[(bnseg + 4*i + 2) * BSTR + bk] = f32_to_bf16(fb[i].z);
      Bls[(bnseg + 4*i + 3) * BSTR + bk] = f32_to_bf16(fb[i].w);
    }
    // ---- prefetch next K-step while this one computes ----
    if (kk + BK < K) load_globals(kk + BK);
    __syncthreads();

    v16bf afr[2], bfr[4];
#pragma unroll
    for (int i = 0; i < 2; i++)
      afr[i] = load_a_frag(&Als[(wm + 16*i + frow) * ASTR], akoff);
#pragma unroll
    for (int j = 0; j < 4; j++)
      bfr[j] = load_b_frag(&Bls[(wn + 16*j + frow) * BSTR], bkbase);
#pragma unroll
    for (int i = 0; i < 2; i++)
#pragma unroll
      for (int j = 0; j < 4; j++)
        acc[i][j] = wmma_bf16(afr[i], bfr[j], acc[i][j]);
  }

  // ---- epilogue ----
  const int col_l  = lane & 15;
  const int rowadd = (lane < 16) ? 0 : 8;
#pragma unroll
  for (int i = 0; i < 2; i++) {
#pragma unroll
    for (int j = 0; j < 4; j++) {
      const int gc = n0 + wn + 16*j + col_l;
      const float bj = bias[gc];
#pragma unroll
      for (int r = 0; r < 8; r++) {
        const int gr = m0 + wm + 16*i + r + rowadd;
        const float v = acc[i][j][r] + bj;
        if (EPI == 0) {
          const int which = gc >> 11;
          const int rem = gc & 2047;
          const int hh = rem >> 7, d = rem & 127;
          const int bidx = gr >> 10, t = gr & 1023;
          const int bh = bidx * NHEADS + hh;
          if (which == 0)
            qout[(size_t)(bh * SEQ + t) * HEADDIM + d] = f32_to_bf16(v * QKV_SCALE);
          else if (which == 1)
            kout[(size_t)(bh * SEQ + t) * HEADDIM + d] = f32_to_bf16(v);
          else   // V stored transposed: [bh][d][t]
            vtout[(size_t)(bh * HEADDIM + d) * SEQ + t] = f32_to_bf16(v);
        } else {
          fout[(size_t)gr * N + gc] = v;
        }
      }
    }
  }
}

// ---------------------------------------------------------------------------
// Flash attention (causal): one (b,h) x 128-query tile per workgroup.
// 8 waves x 16 queries; Q in registers as A-fragments.
// K / V^T tiles double-buffered in LDS, staged by the Tensor Data Mover
// (wave 0 issues descriptors, TENSORcnt-waits, barrier publishes).
// ---------------------------------------------------------------------------
__global__ __launch_bounds__(256) void attn_wmma_kernel(
    const unsigned short* __restrict__ Q, const unsigned short* __restrict__ Kt,
    const unsigned short* __restrict__ Vt, unsigned short* __restrict__ aout)
{
  constexpr int KSTR = 136, VSTR = 72, PSTR = 72;   // padded LDS strides (elems)
  constexpr int KBUF = 64 * KSTR, VBUF = 128 * VSTR;
  __shared__ __align__(16) unsigned short Kls[2 * KBUF];     // [buf][key][d]
  __shared__ __align__(16) unsigned short Vls[2 * VBUF];     // [buf][d][key]
  __shared__ __align__(16) unsigned short Pls[8 * 16 * PSTR];// per-wave [m][k]

  const int tid  = threadIdx.x;
  const int lane = tid & 31;
  const int wave = tid >> 5;
  const int bh = blockIdx.x >> 3;
  const int q0 = (blockIdx.x & 7) * 128;

  const unsigned short* Qb = Q  + (size_t)bh * SEQ * HEADDIM;
  const unsigned short* Kb = Kt + (size_t)bh * SEQ * HEADDIM;
  const unsigned short* Vb = Vt + (size_t)bh * HEADDIM * SEQ;

  const int row_l  = lane & 15;
  const int half   = (lane < 16) ? 0 : 1;
  const int akoff  = half ? 8 : 0;
  const int bkbase = half ? 8 : 0;

  // Q fragments: 4 k-steps over HEADDIM (pre-scaled by 1/sqrt(Dh))
  v16bf qf[4];
  {
    const unsigned short* qrow = Qb + (size_t)(q0 + wave*16 + row_l) * HEADDIM;
#pragma unroll
    for (int s = 0; s < 4; s++) qf[s] = load_a_frag(qrow, 32*s + akoff);
  }

  v8f oacc[8];
#pragma unroll
  for (int dt = 0; dt < 8; dt++) oacc[dt] = zero_v8f();
  float mrow[8], lrow[8];
#pragma unroll
  for (int r = 0; r < 8; r++) { mrow[r] = -3.0e38f; lrow[r] = 0.0f; }

  unsigned short* Pw = Pls + wave * 16 * PSTR;
  const int nkb = q0 / 64 + 2;                 // causal: keys <= q0+127

#if HAS_TDM
  // prologue: DMA block 0 into buffer 0 (K: rows of 128 elems -> pad codes 5/3;
  // V^T: rows of 64 elems -> pad codes 4/3; both add 4 dwords -> strides 136/72)
  if (wave == 0) {
    tdm_load_2d_bf16(Kb, lds_byte_off(&Kls[0]), HEADDIM, 64, HEADDIM, 5, 3);
    tdm_load_2d_bf16(Vb, lds_byte_off(&Vls[0]), 64, HEADDIM, SEQ,     4, 3);
  }
#endif

  for (int kb = 0; kb < nkb; kb++) {
    const int kb0 = kb * 64;
    const int cur = kb & 1;
    unsigned short* Kcur = Kls + cur * KBUF;
    unsigned short* Vcur = Vls + cur * VBUF;

    __syncthreads();   // everyone done reading buffer cur^1 from previous block
#if HAS_TDM
    if (wave == 0) {
      if (kb + 1 < nkb) {
        const int nb0 = kb0 + 64;
        tdm_load_2d_bf16(Kb + (size_t)nb0 * HEADDIM,
                         lds_byte_off(&Kls[(cur ^ 1) * KBUF]), HEADDIM, 64, HEADDIM, 5, 3);
        tdm_load_2d_bf16(Vb + nb0,
                         lds_byte_off(&Vls[(cur ^ 1) * VBUF]), 64, HEADDIM, SEQ, 4, 3);
        __builtin_amdgcn_s_wait_tensorcnt(2);  // in-order: current pair complete
      } else {
        __builtin_amdgcn_s_wait_tensorcnt(0);
      }
    }
#else
    { // manual staging fallback
      const int krow = tid >> 2, kseg = (tid & 3) * 32;
      const unsigned short* src = Kb + (size_t)(kb0 + krow) * HEADDIM + kseg;
      uint4* dst = (uint4*)&Kcur[krow * KSTR + kseg];
#pragma unroll
      for (int i = 0; i < 4; i++) dst[i] = ((const uint4*)src)[i];
      const int vrow = tid >> 1, vseg = (tid & 1) * 32;
      const unsigned short* vsrc = Vb + (size_t)vrow * SEQ + kb0 + vseg;
      uint4* vdst = (uint4*)&Vcur[vrow * VSTR + vseg];
#pragma unroll
      for (int i = 0; i < 4; i++) vdst[i] = ((const uint4*)vsrc)[i];
    }
#endif
    __syncthreads();   // publish buffer cur

    // ---- S = Q K^T : 16x64 strip per wave ----
    v8f sacc[4];
#pragma unroll
    for (int j = 0; j < 4; j++) sacc[j] = zero_v8f();
#pragma unroll
    for (int j = 0; j < 4; j++)
#pragma unroll
      for (int s = 0; s < 4; s++) {
        v16bf kf = load_b_frag(&Kcur[(16*j + row_l) * KSTR], 32*s + bkbase);
        sacc[j] = wmma_bf16(qf[s], kf, sacc[j]);
      }

    // ---- causal mask (trailing blocks only) ----
    if (kb0 + 63 > q0 + wave * 16) {
#pragma unroll
      for (int j = 0; j < 4; j++) {
        const int key = kb0 + 16*j + row_l;
#pragma unroll
        for (int r = 0; r < 8; r++) {
          const int q = q0 + wave*16 + r + 8*half;
          if (key > q) sacc[j][r] = -3.0e38f;
        }
      }
    }

    // ---- online softmax (rows spread over 16 lanes) ----
#pragma unroll
    for (int r = 0; r < 8; r++) {
      float rmax = fmaxf(fmaxf(sacc[0][r], sacc[1][r]), fmaxf(sacc[2][r], sacc[3][r]));
#pragma unroll
      for (int off = 8; off > 0; off >>= 1)
        rmax = fmaxf(rmax, __shfl_xor(rmax, off, 16));
      const float mnew  = fmaxf(mrow[r], rmax);
      const float alpha = __expf(mrow[r] - mnew);
      float rsum = 0.0f;
#pragma unroll
      for (int j = 0; j < 4; j++) {
        const float p = __expf(sacc[j][r] - mnew);
        sacc[j][r] = p;
        rsum += p;
      }
#pragma unroll
      for (int off = 8; off > 0; off >>= 1)
        rsum += __shfl_xor(rsum, off, 16);
      lrow[r] = lrow[r] * alpha + rsum;
      mrow[r] = mnew;
#pragma unroll
      for (int dt = 0; dt < 8; dt++) oacc[dt][r] *= alpha;
    }

    // ---- P (D-layout) -> wave-private LDS, reload as A-fragments ----
#pragma unroll
    for (int j = 0; j < 4; j++)
#pragma unroll
      for (int r = 0; r < 8; r++)
        Pw[(r + 8*half) * PSTR + 16*j + row_l] = f32_to_bf16(sacc[j][r]);
    asm volatile("s_wait_dscnt 0" ::: "memory");   // wave-internal LDS RAW

    v16bf pf[2];
#pragma unroll
    for (int s2 = 0; s2 < 2; s2++)
      pf[s2] = load_a_frag(&Pw[row_l * PSTR], 32*s2 + akoff);

    // ---- O += P * V ----
#pragma unroll
    for (int dt = 0; dt < 8; dt++)
#pragma unroll
      for (int s2 = 0; s2 < 2; s2++) {
        v16bf vf = load_b_frag(&Vcur[(16*dt + row_l) * VSTR], 32*s2 + bkbase);
        oacc[dt] = wmma_bf16(pf[s2], vf, oacc[dt]);
      }
  }

  // ---- normalize and write attn_out bf16 [B*T][DMODEL] ----
  const int b = bh >> 4, h = bh & 15;
#pragma unroll
  for (int r = 0; r < 8; r++) {
    const float rl = 1.0f / lrow[r];
    const int t = q0 + wave*16 + r + 8*half;
    unsigned short* orow = aout + (size_t)(b * SEQ + t) * DMODEL + h * HEADDIM;
#pragma unroll
    for (int dt = 0; dt < 8; dt++)
      orow[16*dt + row_l] = f32_to_bf16(oacc[dt][r] * rl);
  }
}

// ---------------------------------------------------------------------------
// Launch: GEMM1 (QKV, scatter) -> flash attention -> GEMM2 (output proj)
// Workspace: Q | K | V^T | attn_out (bf16, 16 MB each = 64 MB)
// ---------------------------------------------------------------------------
extern "C" void kernel_launch(void* const* d_in, const int* in_sizes, int n_in,
                              void* d_out, int out_size, void* d_ws, size_t ws_size,
                              hipStream_t stream) {
  const float* x     = (const float*)d_in[0];
  const float* w_qkv = (const float*)d_in[1];
  const float* b_qkv = (const float*)d_in[2];
  const float* w_out = (const float*)d_in[3];
  const float* b_out = (const float*)d_in[4];
  float* out = (float*)d_out;

  const size_t TENS = (size_t)BATCH * NHEADS * SEQ * HEADDIM;   // 8,388,608 elems
  unsigned short* qws  = (unsigned short*)d_ws;
  unsigned short* kws  = qws  + TENS;
  unsigned short* vtws = kws  + TENS;
  unsigned short* aows = vtws + TENS;

  dim3 g1(3 * DMODEL / 128, MROWS / 128);
  gemm_wmma_kernel<0><<<g1, 256, 0, stream>>>(
      x, nullptr, w_qkv, b_qkv, DMODEL, 3 * DMODEL,
      qws, kws, vtws, nullptr);

  attn_wmma_kernel<<<dim3(BATCH * NHEADS * (SEQ / 128)), 256, 0, stream>>>(
      qws, kws, vtws, aows);

  dim3 g2(DMODEL / 128, MROWS / 128);
  gemm_wmma_kernel<1><<<g2, 256, 0, stream>>>(
      nullptr, aows, w_out, b_out, DMODEL, DMODEL,
      nullptr, nullptr, nullptr, out);
}
